// WordAttention_72095321030959
// MI455X (gfx1250) — matrix operand order repaired and checked by reading, not compiled, not run on hardware
//
#include <hip/hip_runtime.h>
#include <cstdint>

// Problem constants (from reference): B=16, S=32, TK=128, N=512, L=S*TK=4096
#define BB 16
#define SS 32
#define TKK 128
#define NN 512
#define LL 4096

typedef __attribute__((ext_vector_type(4)))  unsigned u32x4;
typedef __attribute__((ext_vector_type(4)))  int      i32x4;
typedef __attribute__((ext_vector_type(4)))  float    f32x4;
typedef __attribute__((ext_vector_type(8)))  float    v8f;
typedef __attribute__((ext_vector_type(16))) __bf16   v16bf;

#if __has_builtin(__builtin_amdgcn_global_load_async_to_lds_b128) && \
    __has_builtin(__builtin_amdgcn_s_wait_asynccnt)
#define USE_ASYNC_LDS 1
#endif

#ifdef USE_ASYNC_LDS
// 16B global -> LDS async copy (ASYNCcnt-tracked). AMDGPU addrspacecasts:
// generic->global is the identity on the 64-bit address; generic->local is a
// truncation to addr[31:0] (ISA 10.2 aperture rules), so integer detours are
// bit-exact and always compile.
static __device__ inline void async_copy16(const float* g, float* l) {
    __attribute__((address_space(1))) i32x4* gp =
        (__attribute__((address_space(1))) i32x4*)(uintptr_t)g;
    __attribute__((address_space(3))) i32x4* lp =
        (__attribute__((address_space(3))) i32x4*)(uint32_t)(uintptr_t)l;
    __builtin_amdgcn_global_load_async_to_lds_b128(gp, lp, 0, 0);
}
#endif

struct FragBits { u32x4 lo, hi; };

static __device__ inline v16bf make_frag(u32x4 lo, u32x4 hi) {
    FragBits p{lo, hi};
    return __builtin_bit_cast(v16bf, p);
}

// round-to-nearest-even fp32 -> packed 2x bf16
static __device__ inline unsigned bf16pk(float flo, float fhi) {
    unsigned a = __float_as_uint(flo), b = __float_as_uint(fhi);
    a = (a + 0x7FFFu + ((a >> 16) & 1u)) >> 16;
    b = (b + 0x7FFFu + ((b >> 16) & 1u)) >> 16;
    return (a & 0xFFFFu) | (b << 16);
}

// ---------------------------------------------------------------------------
// K0a: pre-swizzle W_h (N x N fp32, row-major W_h[j][k]) into the WMMA
// B operand (32x16 bf16, K x N) in this code's contiguous K order (not AMD's table order):
//   B element (k, j): lane = (k/16 selects half)*16 + (j mod 16)  [N = lane&15]
//   VGPR i holds K = (lane>>4)*16 + {2i, 2i+1}  packed [15:0],[31:16]
// Output index: Wb[ ((jt*16 + kt)*32 + lane)*8 + i ]
// ---------------------------------------------------------------------------
__global__ __launch_bounds__(256) void k0_pack(const float* __restrict__ Wh,
                                               unsigned* __restrict__ Wb) {
    int o = blockIdx.x * 256 + threadIdx.x;      // 32*16*32*8 = 131072 entries
    int i    = o & 7;
    int lane = (o >> 3) & 31;
    int rest = o >> 8;
    int kt   = rest & 15;
    int jt   = rest >> 4;
    int j = jt * 16 + (lane & 15);
    int k = kt * 32 + (lane >> 4) * 16 + 2 * i;
    Wb[o] = bf16pk(Wh[(size_t)j * NN + k], Wh[(size_t)j * NN + k + 1]);
}

// ---------------------------------------------------------------------------
// K0b: dec_fea[b][n] = sum_k s_t_hat[b][k] * W_d[n][k] + b_d[n]   (fp32, tiny)
// ---------------------------------------------------------------------------
__global__ __launch_bounds__(256) void k0_dec(const float* __restrict__ s_t_hat,
                                              const float* __restrict__ Wd,
                                              const float* __restrict__ bd,
                                              float* __restrict__ dec) {
    int o = blockIdx.x * 256 + threadIdx.x;      // B*N = 8192
    int b = o >> 9;
    int n = o & (NN - 1);
    const float* sb = s_t_hat + (size_t)b * NN;
    const float* wr = Wd + (size_t)n * NN;
    float a = bd[n];
#pragma unroll 8
    for (int k = 0; k < NN; ++k) a += sb[k] * wr[k];
    dec[o] = a;
}

// ---------------------------------------------------------------------------
// K1: fused scores kernel.
//  grid = B * (L/32) = 2048 blocks, 128 threads (4 waves).
//  Block handles 32 rows (2 WMMA row-tiles). Each wave owns 8 N-tiles.
//  E = Htile(32x512) @ W_h^T  via v_wmma_f32_16x16x32_bf16, then fused
//  tanh(E + dec_fea + cov*W_c) . v  ->  beta-weighted scores.
//  B-fragment stream is software-pipelined (depth 2) so L2 latency overlaps
//  with the WMMAs instead of a full s_wait_loadcnt 0 per step.
// ---------------------------------------------------------------------------
#define ROWS 32
#define LDSW 260   // u32 per LDS row: 512 bf16 data + 8 bf16 pad (bank skew)

__global__ __launch_bounds__(128) void k1_scores(
    const float* __restrict__ h, const float* __restrict__ coverage,
    const float* __restrict__ beta, const float* __restrict__ Wc,
    const float* __restrict__ v, const unsigned* __restrict__ Wb,
    const float* __restrict__ dec_fea, float* __restrict__ scores) {

    __shared__ unsigned Ht[ROWS * LDSW];
    __shared__ float sc[ROWS];

    int tid  = threadIdx.x;
    int bid  = blockIdx.x;
    int b    = bid >> 7;            // 128 tiles per batch
    int l0   = (bid & 127) * ROWS;

    // ---- cooperative load 32x512 fp32 tile -> bf16 LDS (row-major, padded)
    const f32x4* hb = (const f32x4*)(h + ((size_t)b * LL + l0) * NN);
#pragma unroll
    for (int it = 0; it < 32; ++it) {
        int idx = tid + it * 128;            // 4096 float4 total
        int row = idx >> 7;                  // 128 float4 per row
        int c4  = idx & 127;
        f32x4 f = hb[(size_t)row * 128 + c4];
        unsigned* dst = &Ht[row * LDSW + c4 * 2];
        dst[0] = bf16pk(f.x, f.y);
        dst[1] = bf16pk(f.z, f.w);
    }
    if (tid < ROWS) sc[tid] = 0.0f;
    __syncthreads();

    int lane  = tid & 31;
    int wave  = tid >> 5;
    int khalf = lane >> 4;     // K-half for A/B fragments
    int nlo   = lane & 15;     // M row (A), N col (B/C)

    // hoist per-row coverage (C-matrix row layout: VGPR r -> M = r + khalf*8)
    float covv[2][8];
#pragma unroll
    for (int rt = 0; rt < 2; ++rt)
#pragma unroll
        for (int r = 0; r < 8; ++r)
            covv[rt][r] = coverage[(size_t)b * LL + l0 + rt * 16 + khalf * 8 + r];

    float acc[2][8] = {};

    for (int jj = 0; jj < 8; ++jj) {
        int jt = wave * 8 + jj;
        int j  = jt * 16 + nlo;
        v8f C0 = {}; v8f C1 = {};

        // B fragment pointer for (jt, kt): 8 dwords per lane, coalesced 1KB/wave
        const u32x4* bbase =
            (const u32x4*)(Wb + (((jt * 16) * 32 + lane) << 3));
        // per-kt stride in u32x4 units: 32 lanes * 8 dwords = 256 dwords = 64
        // depth-2 software pipeline: keep 2 B fragments in flight
        u32x4 bl0 = bbase[0],      bh0 = bbase[1];        // kt = 0
        u32x4 bl1 = bbase[64],     bh1 = bbase[64 + 1];   // kt = 1
#pragma unroll
        for (int kt = 0; kt < 16; ++kt) {
            u32x4 nbl, nbh;
            if (kt + 2 < 16) {
                nbl = bbase[(kt + 2) * 64];
                nbh = bbase[(kt + 2) * 64 + 1];
            }
            v16bf Bf = make_frag(bl0, bh0);
            // A fragments from LDS: K base = kt*32 + khalf*16, 16 contiguous K (32B)
            int kb = kt * 16 + khalf * 8;    // u32 offset in row
            const u32x4* a0 = (const u32x4*)&Ht[nlo * LDSW + kb];
            const u32x4* a1 = (const u32x4*)&Ht[(16 + nlo) * LDSW + kb];
            v16bf A0 = make_frag(a0[0], a0[1]);
            v16bf A1 = make_frag(a1[0], a1[1]);
            C0 = __builtin_amdgcn_wmma_f32_16x16x32_bf16(false, A0, false, Bf,
                                                         (short)0, C0, false, false);
            C1 = __builtin_amdgcn_wmma_f32_16x16x32_bf16(false, A1, false, Bf,
                                                         (short)0, C1, false, false);
            bl0 = bl1; bh0 = bh1;
            if (kt + 2 < 16) { bl1 = nbl; bh1 = nbh; }
        }
        // fused epilogue for column j
        float dec = dec_fea[b * NN + j];
        float wcj = Wc[j];
        float vj  = v[j];
#pragma unroll
        for (int r = 0; r < 8; ++r) {
            float x0 = C0[r] + dec + covv[0][r] * wcj;
            float x1 = C1[r] + dec + covv[1][r] * wcj;
            acc[0][r] += tanhf(x0) * vj;
            acc[1][r] += tanhf(x1) * vj;
        }
    }

    // reduce over the 16 lanes of each half (columns j within this wave)
#pragma unroll
    for (int m = 1; m < 16; m <<= 1)
#pragma unroll
        for (int rt = 0; rt < 2; ++rt)
#pragma unroll
            for (int r = 0; r < 8; ++r)
                acc[rt][r] += __shfl_xor(acc[rt][r], m, 32);

    if (nlo == 0) {
#pragma unroll
        for (int rt = 0; rt < 2; ++rt)
#pragma unroll
            for (int r = 0; r < 8; ++r)
                atomicAdd(&sc[rt * 16 + khalf * 8 + r], acc[rt][r]);
    }
    __syncthreads();

    if (tid < ROWS) {
        int l = l0 + tid;
        scores[(size_t)b * LL + l] = beta[b * SS + (l >> 7)] * sc[tid];
    }
}

// ---------------------------------------------------------------------------
// K2: per-batch masked softmax + renormalize (Z cancels) + coverage update.
//  out layout: [0,8192) c_t | [8192, 8192+65536) attn | then coverage_new
// ---------------------------------------------------------------------------
__global__ __launch_bounds__(256) void k2_softmax(const float* __restrict__ scores,
                                                  const float* __restrict__ mask,
                                                  const float* __restrict__ coverage,
                                                  float* __restrict__ out) {
    __shared__ float red[256];
    int b = blockIdx.x, tid = threadIdx.x;
    const float* w  = scores + (size_t)b * LL;
    const float* mk = mask   + (size_t)b * LL;

    float mx = -3.4e38f;
    for (int i = tid; i < LL; i += 256) mx = fmaxf(mx, w[i]);
    red[tid] = mx; __syncthreads();
    for (int s = 128; s > 0; s >>= 1) {
        if (tid < s) red[tid] = fmaxf(red[tid], red[tid + s]);
        __syncthreads();
    }
    mx = red[0]; __syncthreads();

    float sum = 0.0f;
    for (int i = tid; i < LL; i += 256) sum += __expf(w[i] - mx) * mk[i];
    red[tid] = sum; __syncthreads();
    for (int s = 128; s > 0; s >>= 1) {
        if (tid < s) red[tid] += red[tid + s];
        __syncthreads();
    }
    float inv = 1.0f / red[0];

    float*       attn = out + BB * NN + (size_t)b * LL;
    float*       covn = out + BB * NN + (size_t)BB * LL + (size_t)b * LL;
    const float* cov  = coverage + (size_t)b * LL;
    for (int i = tid; i < LL; i += 256) {
        float a = __expf(w[i] - mx) * mk[i] * inv;
        attn[i] = a;
        covn[i] = cov[i] + a;
    }
}

// ---------------------------------------------------------------------------
// K3: c_t[b][n] = sum_l attn[b][l] * h[b][l][n]   (memory-bound GEMV)
//  attn tile staged to LDS with GLOBAL_LOAD_ASYNC_TO_LDS_B128 (ASYNCcnt path)
//  when the toolchain exposes the gfx1250 async builtins.
// ---------------------------------------------------------------------------
__global__ __launch_bounds__(256) void k3_ctx(const float* __restrict__ h,
                                              const float* __restrict__ attn_all,
                                              float* __restrict__ out) {
    __shared__ float a[LL];
    int b = blockIdx.x, n0 = blockIdx.y * 256, tid = threadIdx.x;
    const float* attn = attn_all + (size_t)b * LL;
#ifdef USE_ASYNC_LDS
    // 4096 floats = 1024 x 16B transfers; 256 lanes x 4 async b128 each
#pragma unroll
    for (int it = 0; it < 4; ++it) {
        int idx4 = tid + it * 256;           // float4 index
        async_copy16(attn + idx4 * 4, &a[idx4 * 4]);
    }
    __builtin_amdgcn_s_wait_asynccnt(0);
    __syncthreads();
#else
    for (int i = tid; i < LL; i += 256) a[i] = attn[i];
    __syncthreads();
#endif
    const float* hb = h + (size_t)b * LL * NN + n0 + tid;
    float acc = 0.0f;
#pragma unroll 8
    for (int l = 0; l < LL; ++l) acc += a[l] * hb[(size_t)l * NN];
    out[b * NN + n0 + tid] = acc;
}

// ---------------------------------------------------------------------------
extern "C" void kernel_launch(void* const* d_in, const int* in_sizes, int n_in,
                              void* d_out, int out_size, void* d_ws, size_t ws_size,
                              hipStream_t stream) {
    const float* s_t_hat  = (const float*)d_in[0];   // (B,N)
    const float* h        = (const float*)d_in[1];   // (B,S,TK,N)
    const float* coverage = (const float*)d_in[2];   // (B,S,TK)
    const float* mask     = (const float*)d_in[3];   // (B,S,TK)
    const float* beta     = (const float*)d_in[4];   // (B,S)
    const float* Wh       = (const float*)d_in[5];   // (N,N)
    const float* Wc       = (const float*)d_in[6];   // (N,)
    const float* Wd       = (const float*)d_in[7];   // (N,N)
    const float* bd       = (const float*)d_in[8];   // (N,)
    const float* v        = (const float*)d_in[9];   // (N,)
    float* out = (float*)d_out;

    // workspace: Wb (131072 u32) | dec_fea (8192 f32) | scores (65536 f32)
    unsigned* Wb     = (unsigned*)d_ws;
    float*    dec    = (float*)d_ws + 131072;
    float*    scores = (float*)d_ws + 131072 + BB * NN;

    k0_pack<<<512, 256, 0, stream>>>(Wh, Wb);
    k0_dec<<<32, 256, 0, stream>>>(s_t_hat, Wd, bd, dec);
    k1_scores<<<BB * (LL / ROWS), 128, 0, stream>>>(h, coverage, beta, Wc, v,
                                                    Wb, dec, scores);
    k2_softmax<<<BB, 256, 0, stream>>>(scores, mask, coverage, out);
    k3_ctx<<<dim3(BB, 2), 256, 0, stream>>>(h, out + BB * NN, out);
}
